// LSTMAEModule_66005057405444
// MI455X (gfx1250) — compile-verified, matrix operand order checked by reading, not run on hardware
//
#include <hip/hip_runtime.h>
#include <hip/hip_bf16.h>

// ---------------------------------------------------------------------------
// LSTM-AE forward for MI455X (gfx1250, wave32, WMMA).
// B=256, T=256, F=128, H=512. One workgroup per 16 batch rows; recurrence is
// per-row independent so no grid sync. Matmuls in bf16 WMMA w/ f32 accumulate,
// fp32 h/c state in LDS. Weights stream from L2 every step (2.6 MB bf16 is
// L2-resident). An opaque-asm pointer redefinition per timestep blocks LICM
// (which previously hoisted 320 fragment loads and spilled them to scratch);
// the pointers are kept in address_space(1) so the loads lower to
// GLOBAL_LOAD_B128 (LOADcnt only) instead of FLAT (LOADcnt+DScnt).
// ---------------------------------------------------------------------------

typedef __bf16  v16bf __attribute__((ext_vector_type(16)));
typedef float   v8f   __attribute__((ext_vector_type(8)));
typedef int     v4i   __attribute__((ext_vector_type(4)));

typedef __attribute__((address_space(1))) const unsigned short gu16;
typedef __attribute__((address_space(1))) const v4i            gv4i;

#define B_  256
#define T_  256
#define F_  128
#define H_  512
#define G4  (4 * H_)   // 2048 gate rows

// LDS row strides (padded to avoid bank conflicts; keep 16B alignment)
#define HS   516       // f32 h/c row stride (516 % 64 == 4 dwords)
#define HBS  520       // bf16 h row stride  (520 = 8*65 -> 16B-aligned rows)
#define XBS  136       // bf16 x row stride  (136 = 8*17)

__device__ inline v16bf frag_from(const v4i* lo_ptr) {
    union { v4i i[2]; v16bf v; } u;
    u.i[0] = lo_ptr[0];   // K 0..7   (or 8..15 for upper lane half)
    u.i[1] = lo_ptr[2];   // K 16..23 (or 24..31): +16 elements = +2 v4i
    return u.v;
}

__device__ inline v16bf frag_from_g(gv4i* lo_ptr) {
    union { v4i i[2]; v16bf v; } u;
    u.i[0] = lo_ptr[0];
    u.i[1] = lo_ptr[2];
    return u.v;
}

__global__ void lstmae_prep(const float* __restrict__ wih,
                            const float* __restrict__ whh,
                            const float* __restrict__ wout,
                            const float* __restrict__ bih,
                            const float* __restrict__ bhh,
                            __hip_bfloat16* __restrict__ wih_b,
                            __hip_bfloat16* __restrict__ whh_b,
                            __hip_bfloat16* __restrict__ wout_b,
                            float* __restrict__ bias) {
    int i = blockIdx.x * 256 + threadIdx.x;
    const int n_ih  = G4 * F_;   // 262144
    const int n_hh  = G4 * H_;   // 1048576 (grid sized to this)
    const int n_out = F_ * H_;   // 65536
    if (i < n_ih)  wih_b[i]  = __float2bfloat16(wih[i]);
    if (i < n_hh)  whh_b[i]  = __float2bfloat16(whh[i]);
    if (i < n_out) wout_b[i] = __float2bfloat16(wout[i]);
    if (i < G4)    bias[i]   = bih[i] + bhh[i];
}

__global__ __launch_bounds__(256, 1) void lstmae_main(
    const float* __restrict__ ts,                 // [B,T,F]
    const __hip_bfloat16* __restrict__ wih_b,     // [2048,128] row-major
    const __hip_bfloat16* __restrict__ whh_b,     // [2048,512] row-major
    const __hip_bfloat16* __restrict__ wout_b,    // [128,512]  row-major
    const float* __restrict__ bias,               // [2048] (b_ih+b_hh)
    const float* __restrict__ b_out,              // [128]
    float* __restrict__ d_err,                    // [256]
    float* __restrict__ d_outm)                   // [256,128]
{
    __shared__ __align__(16) float          h_s[16 * HS];
    __shared__ __align__(16) float          c_s[16 * HS];
    __shared__ __align__(16) __hip_bfloat16 hb_s[16 * HBS];
    __shared__ __align__(16) __hip_bfloat16 xb_s[16 * XBS];
    __shared__ __align__(16) float          ob_s[16 * F_];

    const int tid   = threadIdx.x;
    const int lane  = tid & 31;
    const int wv    = tid >> 5;            // 0..7
    const int b0    = blockIdx.x * 16;     // batch tile base
    const int lrow  = lane & 15;           // row (A) / col (B,C,D) within tile
    const int khalf = (lane >> 4) << 3;    // 0 or 8: K sub-block per lane half
    const int mbase = (lane >> 4) << 3;    // C/D: M = vgpr + 0 or 8

    for (int i = tid; i < 16 * HS; i += 256) { h_s[i] = 0.f; c_s[i] = 0.f; }
    __syncthreads();

    v8f acc[4][4];                         // [gate][hidden 16-col tile]

#pragma unroll 1
    for (int t = 0; t < T_; ++t) {
        // ---- stage x_t and h as bf16 into LDS -----------------------------
        for (int i = tid; i < 16 * F_; i += 256) {
            int r = i >> 7, c = i & 127;
            xb_s[r * XBS + c] =
                __float2bfloat16(ts[(size_t)(b0 + r) * T_ * F_ + (size_t)t * F_ + c]);
        }
        for (int i = tid; i < 16 * H_; i += 256) {
            int r = i >> 9, c = i & 511;
            hb_s[r * HBS + c] = __float2bfloat16(h_s[r * HS + c]);
        }
        __syncthreads();

        // Opaque per-iteration redefinition of the weight bases (SGPRs,
        // global address space). Blocks LICM so the B-fragment loads stay in
        // the loop as GLOBAL_LOAD_B128 streaming from L2.
        gu16* wih_p = (gu16*)(uintptr_t)wih_b;
        gu16* whh_p = (gu16*)(uintptr_t)whh_b;
        asm volatile("" : "+s"(wih_p), "+s"(whh_p));

        v8f z = {0.f, 0.f, 0.f, 0.f, 0.f, 0.f, 0.f, 0.f};
#pragma unroll
        for (int g = 0; g < 4; ++g)
#pragma unroll
            for (int n = 0; n < 4; ++n) acc[g][n] = z;

        // ---- gates += x_t @ W_ih^T : K = 128 ------------------------------
        for (int kc = 0; kc < 4; ++kc) {
            const int k0 = kc * 32;
            v16bf a = frag_from((const v4i*)(xb_s + lrow * XBS + k0 + khalf));
#pragma unroll
            for (int g = 0; g < 4; ++g)
#pragma unroll
                for (int n = 0; n < 4; ++n) {
                    const int row = g * H_ + wv * 64 + n * 16 + lrow;
                    v16bf b = frag_from_g((gv4i*)(wih_p + (size_t)row * F_ + k0 + khalf));
                    acc[g][n] = __builtin_amdgcn_wmma_f32_16x16x32_bf16(
                        false, a, false, b, (short)0, acc[g][n], false, false);
                }
        }
        // ---- gates += h @ W_hh^T : K = 512 --------------------------------
        for (int kc = 0; kc < 16; ++kc) {
            const int k0 = kc * 32;
            v16bf a = frag_from((const v4i*)(hb_s + lrow * HBS + k0 + khalf));
#pragma unroll
            for (int g = 0; g < 4; ++g)
#pragma unroll
                for (int n = 0; n < 4; ++n) {
                    const int row = g * H_ + wv * 64 + n * 16 + lrow;
                    v16bf b = frag_from_g((gv4i*)(whh_p + (size_t)row * H_ + k0 + khalf));
                    acc[g][n] = __builtin_amdgcn_wmma_f32_16x16x32_bf16(
                        false, a, false, b, (short)0, acc[g][n], false, false);
                }
        }

        // ---- elementwise cell update (PyTorch gate order i,f,g,o) ---------
#pragma unroll
        for (int n = 0; n < 4; ++n) {
            const int col = wv * 64 + n * 16 + lrow;    // hidden unit (N)
            const float bi = bias[0 * H_ + col];
            const float bf = bias[1 * H_ + col];
            const float bg = bias[2 * H_ + col];
            const float bo = bias[3 * H_ + col];
#pragma unroll
            for (int r = 0; r < 8; ++r) {
                const int row = r + mbase;              // batch row (M)
                float iv = 1.f / (1.f + __expf(-(acc[0][n][r] + bi)));
                float fv = 1.f / (1.f + __expf(-(acc[1][n][r] + bf)));
                float e2 = __expf(-2.f * (acc[2][n][r] + bg));
                float gv = (1.f - e2) / (1.f + e2);
                float ov = 1.f / (1.f + __expf(-(acc[3][n][r] + bo)));
                float cn = fv * c_s[row * HS + col] + iv * gv;
                float ec = __expf(-2.f * cn);
                c_s[row * HS + col] = cn;
                h_s[row * HS + col] = ov * (1.f - ec) / (1.f + ec);
            }
        }
        __syncthreads();
    }

    // ---- out = h_T @ W_out^T + b_out : wave wv owns cols [wv*16, wv*16+16)
    for (int i = tid; i < 16 * H_; i += 256) {
        int r = i >> 9, c = i & 511;
        hb_s[r * HBS + c] = __float2bfloat16(h_s[r * HS + c]);
    }
    __syncthreads();

    v8f oacc = {0.f, 0.f, 0.f, 0.f, 0.f, 0.f, 0.f, 0.f};
    {
        gu16* wout_p = (gu16*)(uintptr_t)wout_b;
        for (int kc = 0; kc < 16; ++kc) {
            const int k0 = kc * 32;
            v16bf a = frag_from((const v4i*)(hb_s + lrow * HBS + k0 + khalf));
            const int row = wv * 16 + lrow;             // output feature
            v16bf b = frag_from_g((gv4i*)(wout_p + (size_t)row * H_ + k0 + khalf));
            oacc = __builtin_amdgcn_wmma_f32_16x16x32_bf16(
                false, a, false, b, (short)0, oacc, false, false);
        }
    }
    {
        const int ncol = wv * 16 + lrow;
        const float bo = b_out[ncol];
#pragma unroll
        for (int r = 0; r < 8; ++r) {
            const int row = r + mbase;
            const float v = oacc[r] + bo;
            d_outm[(size_t)(b0 + row) * F_ + ncol] = v;
            ob_s[row * F_ + ncol] = v;
        }
    }
    __syncthreads();

    // ---- rec_error_mean = mean_f |out - ts[:, T-1]| -----------------------
    if (tid < 16) {
        const float* xlast = ts + (size_t)(b0 + tid) * T_ * F_ + (size_t)(T_ - 1) * F_;
        float s = 0.f;
        for (int f = 0; f < F_; ++f) s += fabsf(ob_s[tid * F_ + f] - xlast[f]);
        d_err[b0 + tid] = s * (1.f / (float)F_);
    }
}

extern "C" void kernel_launch(void* const* d_in, const int* in_sizes, int n_in,
                              void* d_out, int out_size, void* d_ws, size_t ws_size,
                              hipStream_t stream) {
    (void)in_sizes; (void)n_in; (void)out_size; (void)ws_size;

    const float* ts   = (const float*)d_in[0];
    const float* wih  = (const float*)d_in[1];
    const float* whh  = (const float*)d_in[2];
    const float* bih  = (const float*)d_in[3];
    const float* bhh  = (const float*)d_in[4];
    // d_in[5..8]: decoder params — dead w.r.t. returned values (DCE'd)
    const float* wout = (const float*)d_in[9];
    const float* bout = (const float*)d_in[10];

    char* ws = (char*)d_ws;
    __hip_bfloat16* wih_b  = (__hip_bfloat16*)ws; ws += (size_t)G4 * F_ * 2;
    __hip_bfloat16* whh_b  = (__hip_bfloat16*)ws; ws += (size_t)G4 * H_ * 2;
    __hip_bfloat16* wout_b = (__hip_bfloat16*)ws; ws += (size_t)F_ * H_ * 2;
    float*          biasp  = (float*)ws;          ws += (size_t)G4 * 4;

    lstmae_prep<<<(G4 * H_) / 256, 256, 0, stream>>>(
        wih, whh, wout, bih, bhh, wih_b, whh_b, wout_b, biasp);

    float* d_err  = (float*)d_out;
    float* d_outm = (float*)d_out + B_;
    lstmae_main<<<B_ / 16, 256, 0, stream>>>(
        ts, wih_b, whh_b, wout_b, biasp, bout, d_err, d_outm);
}
